// ReLU_TCN_62165356642403
// MI455X (gfx1250) — compile-verified, matrix operand order checked by reading, not run on hardware
//
#include <hip/hip_runtime.h>
#include <hip/hip_bf16.h>

typedef __attribute__((ext_vector_type(16))) _Float16 v16h;
typedef __attribute__((ext_vector_type(8)))  _Float16 v8h;
typedef __attribute__((ext_vector_type(8)))  float    v8f;

#define B_    64
#define C_    8
#define T_    2048
#define K_    16
#define NF_   9
#define FEAT_ 144
#define FEATP_ 160          // padded K for layer1 (5 chunks of 32)
#define MLP_  16
#define H1_   64
#define H2_   32
#define W_    1988
#define NROW_ (B_*W_)       // 127232
#define TILES_ ((NROW_+15)/16)  // 7952
#define WPB_  4             // waves per block (128 threads)
#define PI_F  3.14159265358979323846f

// fragment-order packed weights in workspace
#define NFRAG_L1 5
#define NFRAG_L2 4
#define FRAG_DFT0 13        // 2 DFT-basis fragments appended after the 13 MLP frags
#define NFRAG_TOT 15
#define WS_WTS_OFF  0       // W_ floats (softmax weights)
#define WS_FRAG_OFF 8192    // 15*512 halfs

// per-wave LDS slice (halfs):
//   [0,4096)     winbuf [128][32] f16  -> later feat [16][160] f16 -> p/h/o overlays
//   [4096,6656)  rawbuf [128][20] f16  (DFT re/im output)
#define WSLICE_ 6656

// ---- per-lane A-fragment load from LDS (16-bit A layout, wave32):
// lanes 0-15: halves 0..7 = K+0..7,  halves 8..15 = K+16..23   (koff=0)
// lanes 16-31: halves 0..7 = K+8..15, halves 8..15 = K+24..31  (koff=8)
__device__ __forceinline__ v16h load_a_frag(const _Float16* base, int stride,
                                            int m, int koff, int kbase) {
  const _Float16* row = base + m * stride + kbase + koff;
  v8h lo = *(const v8h*)(row);
  v8h hi = *(const v8h*)(row + 16);
  return __builtin_shufflevector(lo, hi, 0,1,2,3,4,5,6,7,8,9,10,11,12,13,14,15);
}

// branch-free atan2 (select-based); |err| ~1e-5 rad, far below f16 storage ulp
__device__ __forceinline__ float fast_atan2(float y, float x) {
  const float ax = fabsf(x), ay = fabsf(y);
  const float mx = fmaxf(ax, ay);
  const float mn = fminf(ax, ay);
  const float a  = mn * __frcp_rn(fmaxf(mx, 1e-30f));
  const float s  = a * a;
  float r = fmaf(s, fmaf(s, fmaf(s, fmaf(s, 0.0208351f, -0.0851330f),
                                  0.1801410f), -0.3302995f), 0.9998660f);
  r = r * a;
  r = (ay > ax) ? (1.57079632679f - r) : r;
  r = (x < 0.f) ? (3.14159265359f - r) : r;
  return (y < 0.f) ? -r : r;
}

// branch-free tanh via hardware exp: 1 - 2/(e^{2x}+1); saturates correctly at +-1
__device__ __forceinline__ float fast_tanh(float x) {
  const float t = __expf(2.f * x);
  return 1.f - 2.f * __frcp_rn(t + 1.f);
}

__global__ __launch_bounds__(128, 1)
void tcn_main(const float* __restrict__ x,
              const float* __restrict__ proj_b,
              const float* __restrict__ b1v,
              const float* __restrict__ b2v,
              const float* __restrict__ out_w,
              const float* __restrict__ out_b,
              const float* __restrict__ wts,
              const v16h* __restrict__ frags,
              float* __restrict__ out) {
  __shared__ _Float16 sm[WPB_ * WSLICE_];
  const int tid  = threadIdx.x;
  const int lane = tid & 31;
  const int wave = tid >> 5;
  const int tile = blockIdx.x * WPB_ + wave;

  _Float16* baseW = sm + wave * WSLICE_;
  _Float16* winW  = baseW;                 // [128][32] f16 (A-layout rows m'=ch*16+pos)
  _Float16* rawW  = baseW + 4096;          // [128][20] f16 (re/im per bin)
  _Float16* featW = baseW;                 // [16][160] f16 (overlays winW once consumed)
  _Float16* pW    = baseW;                 // [16][32] f16
  _Float16* hW    = baseW + 512;           // [16][64] f16
  float*    oW    = (float*)(baseW + 512); // [16][32] f32

  const int mrow   = lane & 15;
  const int koff   = (lane < 16) ? 0 : 8;  // A/B fragment K sub-offset
  const int rowoff = (lane < 16) ? 0 : 8;  // C/D row offset

  const int g    = tile * 16 + mrow;
  const int brow = g / W_;
  const int wrow = g - brow * W_;

  // ---------------- stage dilated windows into LDS (f16, K padded 16->32)
  #pragma unroll
  for (int t = 0; t < 4; ++t) {
    const int ch = (lane >> 4) + 2 * t;   // lanes 0-15: ch 0,2,4,6 ; 16-31: 1,3,5,7
    v8h lo, hi, zv;
    if (g < NROW_) {
      const float* xp = x + ((size_t)brow * C_ + ch) * T_ + wrow;
      #pragma unroll
      for (int k = 0; k < 8; ++k) { lo[k] = (_Float16)xp[4*k]; hi[k] = (_Float16)xp[4*(k+8)]; }
    } else {
      #pragma unroll
      for (int k = 0; k < 8; ++k) { lo[k] = (_Float16)0.f; hi[k] = (_Float16)0.f; }
    }
    #pragma unroll
    for (int k = 0; k < 8; ++k) zv[k] = (_Float16)0.f;
    _Float16* dst = winW + (ch * 16 + mrow) * 32;
    *(v8h*)(dst)      = lo;
    *(v8h*)(dst + 8)  = hi;
    *(v8h*)(dst + 16) = zv;
    *(v8h*)(dst + 24) = zv;
  }

  // ---------------- 16-pt rDFT as WMMA: win[16pos x 16tap] @ basis[16tap x 18]
  {
    v16h aw[8];
    #pragma unroll
    for (int ch = 0; ch < 8; ++ch)
      aw[ch] = load_a_frag(winW + ch * 512, 32, mrow, koff, 0);  // winW fully consumed here
    const v16h db0 = frags[(FRAG_DFT0 + 0) * 32 + lane];
    const v16h db1 = frags[(FRAG_DFT0 + 1) * 32 + lane];
    #pragma unroll
    for (int ch = 0; ch < 8; ++ch) {
      v8f z = {};
      v8f d0 = __builtin_amdgcn_wmma_f32_16x16x32_f16(false, aw[ch], false, db0,
                                                      (short)0, z, false, false);
      #pragma unroll
      for (int r = 0; r < 8; ++r)
        rawW[(ch * 16 + r + rowoff) * 20 + mrow] = (_Float16)d0[r];
      v8f d1 = __builtin_amdgcn_wmma_f32_16x16x32_f16(false, aw[ch], false, db1,
                                                      (short)0, z, false, false);
      if (mrow < 4) {   // only output cols 16..17 (bin f=8) matter
        #pragma unroll
        for (int r = 0; r < 8; ++r)
          rawW[(ch * 16 + r + rowoff) * 20 + 16 + mrow] = (_Float16)d1[r];
      }
    }
  }

  // ---------------- log-magnitude & phase -> feature tile [16][160]
  #pragma unroll
  for (int t = 0; t < 4; ++t) {
    const int ch = (lane >> 4) + 2 * t;
    const _Float16* src = rawW + (ch * 16 + mrow) * 20;
    _Float16* dst = featW + mrow * FEATP_ + ch * (NF_ * 2);
    #pragma unroll
    for (int f = 0; f < NF_; ++f) {
      const float re = (float)src[2 * f];
      const float im = (float)src[2 * f + 1];
      const float mag = __logf(1.f + sqrtf(re * re + im * im));
      const float ph  = fast_atan2(im, re) * (1.0f / PI_F);
      dst[2 * f]     = (_Float16)mag;
      dst[2 * f + 1] = (_Float16)ph;
    }
  }
  { // zero pad feature cols 144..159
    _Float16* z = featW + mrow * FEATP_ + FEAT_ + (lane >> 4) * 8;
    #pragma unroll
    for (int j = 0; j < 8; ++j) z[j] = (_Float16)0.f;
  }

  // ---------------- layer 1: [16 x 160] @ [160 x 16], tanh()*pi
  v16h a1[NFRAG_L1];
  #pragma unroll
  for (int c = 0; c < NFRAG_L1; ++c)
    a1[c] = load_a_frag(featW, FEATP_, mrow, koff, c * 32);
  v8f acc = {};
  #pragma unroll
  for (int c = 0; c < NFRAG_L1; ++c) {
    v16h b = frags[c * 32 + lane];
    acc = __builtin_amdgcn_wmma_f32_16x16x32_f16(false, a1[c], false, b,
                                                 (short)0, acc, false, false);
  }
  {
    const float pb = proj_b[mrow];     // D col = lane&15
    #pragma unroll
    for (int r = 0; r < 8; ++r) {
      const int row = r + rowoff;
      const float v = fast_tanh(acc[r] + pb) * PI_F;
      pW[row * 32 + mrow]      = (_Float16)v;
      pW[row * 32 + 16 + mrow] = (_Float16)0.f;  // zero-pad K 16..31
    }
  }

  // ---------------- layer 2: [16 x 32pad] @ [32 x 64], relu
  {
    const v16h a2 = load_a_frag(pW, 32, mrow, koff, 0);
    #pragma unroll
    for (int nt = 0; nt < 4; ++nt) {
      const v16h b = frags[(NFRAG_L1 + nt) * 32 + lane];
      v8f z = {};
      v8f d = __builtin_amdgcn_wmma_f32_16x16x32_f16(false, a2, false, b,
                                                     (short)0, z, false, false);
      const int col = nt * 16 + mrow;
      const float bb = b1v[col];
      #pragma unroll
      for (int r = 0; r < 8; ++r) {
        const float v = d[r] + bb;
        hW[(r + rowoff) * 64 + col] = (_Float16)(v > 0.f ? v : 0.f);
      }
    }
  }

  // ---------------- layer 3: [16 x 64] @ [64 x 32], relu -> f32 LDS
  {
    const v16h a30 = load_a_frag(hW, 64, mrow, koff, 0);
    const v16h a31 = load_a_frag(hW, 64, mrow, koff, 32);
    #pragma unroll
    for (int nt = 0; nt < 2; ++nt) {
      const v16h bk0 = frags[(NFRAG_L1 + NFRAG_L2 + nt * 2 + 0) * 32 + lane];
      const v16h bk1 = frags[(NFRAG_L1 + NFRAG_L2 + nt * 2 + 1) * 32 + lane];
      v8f d = {};
      d = __builtin_amdgcn_wmma_f32_16x16x32_f16(false, a30, false, bk0,
                                                 (short)0, d, false, false);
      d = __builtin_amdgcn_wmma_f32_16x16x32_f16(false, a31, false, bk1,
                                                 (short)0, d, false, false);
      const int col = nt * 16 + mrow;
      const float bb = b2v[col];
      #pragma unroll
      for (int r = 0; r < 8; ++r) {
        const float v = d[r] + bb;
        oW[(r + rowoff) * 32 + col] = (v > 0.f ? v : 0.f);   // stores after all hW reads
      }
    }
  }

  // ---------------- output head + softmax-weighted accumulation
  if (lane < 16) {
    if (g < NROW_) {
      float s = out_b[0];
      #pragma unroll
      for (int n = 0; n < 32; ++n) s += oW[lane * 32 + n] * out_w[n];
      atomicAdd(&out[brow], s * wts[wrow]);
    }
  }
}

// ---- prep: pre-swizzle weights + DFT basis into WMMA B-fragment order (f16).
// B fragment layout (16-bit, wave32): lane holds col n=lane&15;
// halves 0..7 = K+koff+0..7, halves 8..15 = K+koff+16..23, koff=(lane<16?0:8).
__global__ void pack_weights(const float* __restrict__ proj_w,
                             const float* __restrict__ w1,
                             const float* __restrict__ w2,
                             _Float16* __restrict__ frags) {
  const int i = blockIdx.x * blockDim.x + threadIdx.x;
  if (i >= NFRAG_TOT * 512) return;
  const int f    = i >> 9;
  const int lane = (i >> 4) & 31;
  const int j    = i & 15;
  const int nloc = lane & 15;
  const int koff = (lane < 16) ? 0 : 8;
  const int kk   = ((j < 8) ? j : j + 8) + koff;
  float v = 0.f;
  if (f < NFRAG_L1) {                       // layer1: B[k][n] = proj_w[n][k], K pad 144->160
    const int k = f * 32 + kk;
    if (k < FEAT_) v = proj_w[nloc * FEAT_ + k];
  } else if (f < NFRAG_L1 + NFRAG_L2) {     // layer2: B[k][n] = w1[n][k], K pad 16->32
    const int nt = f - NFRAG_L1;
    if (kk < MLP_) v = w1[(nt * 16 + nloc) * MLP_ + kk];
  } else if (f < FRAG_DFT0) {               // layer3: B[k][n] = w2[n][k], K=64 (2 chunks)
    const int idx = f - NFRAG_L1 - NFRAG_L2;
    const int nt = idx >> 1, kc = idx & 1;
    v = w2[(nt * 16 + nloc) * H1_ + kc * 32 + kk];
  } else {                                  // DFT basis: col c=2f+s; re: cos, im: -sin
    const int c = (f - FRAG_DFT0) * 16 + nloc;
    if (kk < K_ && c < NF_ * 2) {
      const int fr = c >> 1;
      const float ang = PI_F * (float)(fr * kk) / 8.0f;
      v = (c & 1) ? -sinf(ang) : cosf(ang);
    }
  }
  frags[i] = (_Float16)v;
}

// ---- prep: softmax(agg_w) into ws, zero d_out for atomic accumulation
__global__ void prep_softmax(const float* __restrict__ agg_w,
                             float* __restrict__ wts,
                             float* __restrict__ out) {
  __shared__ float red[256];
  const int tid = threadIdx.x;
  float m = -3.4e38f;
  for (int i = tid; i < W_; i += 256) m = fmaxf(m, agg_w[i]);
  red[tid] = m; __syncthreads();
  for (int s = 128; s > 0; s >>= 1) {
    if (tid < s) red[tid] = fmaxf(red[tid], red[tid + s]);
    __syncthreads();
  }
  m = red[0]; __syncthreads();
  float s = 0.f;
  for (int i = tid; i < W_; i += 256) s += __expf(agg_w[i] - m);
  red[tid] = s; __syncthreads();
  for (int t = 128; t > 0; t >>= 1) {
    if (tid < t) red[tid] += red[tid + t];
    __syncthreads();
  }
  const float inv = __frcp_rn(red[0]);
  for (int i = tid; i < W_; i += 256) wts[i] = __expf(agg_w[i] - m) * inv;
  if (tid < B_) out[tid] = 0.f;
}

extern "C" void kernel_launch(void* const* d_in, const int* in_sizes, int n_in,
                              void* d_out, int out_size, void* d_ws, size_t ws_size,
                              hipStream_t stream) {
  const float* x      = (const float*)d_in[0];
  const float* proj_w = (const float*)d_in[1];
  const float* proj_b = (const float*)d_in[2];
  const float* w1     = (const float*)d_in[3];
  const float* b1     = (const float*)d_in[4];
  const float* w2     = (const float*)d_in[5];
  const float* b2     = (const float*)d_in[6];
  const float* out_w  = (const float*)d_in[7];
  const float* out_b  = (const float*)d_in[8];
  const float* agg_w  = (const float*)d_in[9];
  float* out = (float*)d_out;

  float*    wts   = (float*)((char*)d_ws + WS_WTS_OFF);
  _Float16* frags = (_Float16*)((char*)d_ws + WS_FRAG_OFF);

  prep_softmax<<<1, 256, 0, stream>>>(agg_w, wts, out);
  pack_weights<<<(NFRAG_TOT * 512 + 255) / 256, 256, 0, stream>>>(proj_w, w1, w2, frags);

  const int blocks = (TILES_ + WPB_ - 1) / WPB_;   // 1988
  tcn_main<<<blocks, 128, 0, stream>>>(x, proj_b, b1, b2, out_w, out_b, wts,
                                       (const v16h*)frags, out);
}
